// TimeSeriesTransformerWithAttn_81028853006918
// MI455X (gfx1250) — matrix operand order repaired and checked by reading, not compiled, not run on hardware
//
#include <hip/hip_runtime.h>

// ---------------------------------------------------------------------------
// TimeSeriesTransformerWithAttn forward on gfx1250 (MI455X, wave32, WMMA)
// f16 WMMA (v_wmma_f32_16x16x32_f16) for all GEMMs + flash attention,
// f32 residual/LN stream. Round 3: sched_barrier-pinned load issue so the
// scheduler cannot sink double-buffer / V-fragment loads down to their uses
// (round-2 asm showed V loads re-serialized with s_wait_loadcnt 0x0).
// ---------------------------------------------------------------------------

typedef __attribute__((ext_vector_type(16))) _Float16 v16h;
typedef __attribute__((ext_vector_type(8)))  _Float16 v8h;
typedef __attribute__((ext_vector_type(8)))  float    v8f;

#define S_    2048
#define B_    4
#define IND_  64
#define D_    512
#define H_    8
#define HD_   64
#define L_    4
#define FF_   2048
#define OUTD_ 64
#define MTOK  (S_*B_)   // 8192 tokens

union FragH { v16h v; v8h h[2]; };

__device__ __forceinline__ v8f wmma_f16(const v16h& a, const v16h& b, v8f c) {
    // D = A(16x32 f16) x B(32x16 f16) + C(16x16 f32)
    return __builtin_amdgcn_wmma_f32_16x16x32_f16(
        /*neg_a=*/false, a, /*neg_b=*/false, b,
        /*c_mod=*/(short)0, c, /*reuse_a=*/false, /*reuse_b=*/false);
}

// Full scheduling fence: nothing may be reordered across this point.
__device__ __forceinline__ void sched_fence() {
    __builtin_amdgcn_sched_barrier(0);
}

// ---------------------------------------------------------------------------
// Generic GEMM: C[M,N] = A[M,K](f16) * W[N,K](f16)^T + bias[N]
// Block: 256 threads = 8 waves (4 M-waves x 2 N-waves).
// Wave tile 64(M) x 32(N); block tile 256 x 64. All shapes divide exactly.
// K-loop is register double-buffered; a sched_barrier between the k+1 load
// block and the k WMMA block guarantees the loads stay hoisted, overlapping
// one full K-step of global latency with matrix math.
// MODE 0: f32 out. MODE 1: relu -> f16. MODE 2: +posenc, f32+f16 out.
// MODE 3: QKV scatter (Q scaled, V transposed).
// ---------------------------------------------------------------------------
template<int MODE>
__global__ void __launch_bounds__(256)
gemm_kernel(const _Float16* __restrict__ A, const _Float16* __restrict__ W,
            const float* __restrict__ bias,
            float* __restrict__ Cf, _Float16* __restrict__ Ch,
            _Float16* __restrict__ Qh, _Float16* __restrict__ Kh,
            _Float16* __restrict__ Vt,
            int N, int K)
{
    const int lane = threadIdx.x & 31;
    const int wave = threadIdx.x >> 5;
    const int ll   = lane & 15;
    const int g    = lane >> 4;
    const int m0   = blockIdx.x * 256 + (wave >> 1) * 64;
    const int n0   = blockIdx.y * 64  + (wave &  1) * 32;

    // Per-lane row base pointers (K-offset added via immediate/loop offset).
    const _Float16* arow[4];
    #pragma unroll
    for (int i = 0; i < 4; i++)
        arow[i] = A + (size_t)(m0 + i*16 + ll) * K + g*16;
    const _Float16* brow[2];
    #pragma unroll
    for (int j = 0; j < 2; j++)
        brow[j] = W + (size_t)(n0 + j*16 + ll) * K + g*16;

    v8f acc[4][2];
    #pragma unroll
    for (int i = 0; i < 4; i++)
        #pragma unroll
        for (int j = 0; j < 2; j++)
            #pragma unroll
            for (int e = 0; e < 8; e++) acc[i][j][e] = 0.0f;

    FragH a[2][4];
    v16h  b[2][2];

    // prologue: load k0 = 0 fragments into buffer 0
    #pragma unroll
    for (int i = 0; i < 4; i++) {
        a[0][i].h[0] = *(const v8h*)(arow[i]);
        a[0][i].h[1] = *(const v8h*)(arow[i] + 8);
    }
    #pragma unroll
    for (int j = 0; j < 2; j++)
        b[0][j] = *(const v16h*)(brow[j]);

    int cur = 0;
    for (int k0 = 0; k0 < K; k0 += 32) {
        const int nxt = cur ^ 1;
        if (k0 + 32 < K) {
            const int kn = k0 + 32;
            #pragma unroll
            for (int i = 0; i < 4; i++) {
                a[nxt][i].h[0] = *(const v8h*)(arow[i] + kn);
                a[nxt][i].h[1] = *(const v8h*)(arow[i] + kn + 8);
            }
            #pragma unroll
            for (int j = 0; j < 2; j++)
                b[nxt][j] = *(const v16h*)(brow[j] + kn);
        }
        sched_fence();   // keep k+1 loads issued before k's WMMAs
        #pragma unroll
        for (int i = 0; i < 4; i++)
            #pragma unroll
            for (int j = 0; j < 2; j++)
                acc[i][j] = wmma_f16(a[cur][i].v, b[cur][j], acc[i][j]);
        cur = nxt;
    }

    // Epilogue. C layout: lane ll holds col n0+j*16+ll; vgpr r holds row m0+i*16+8*g+r.
    #pragma unroll
    for (int i = 0; i < 4; i++)
        #pragma unroll
        for (int j = 0; j < 2; j++) {
            const int col = n0 + j*16 + ll;
            const float bv = bias[col];
            #pragma unroll
            for (int r = 0; r < 8; r++) {
                const int row = m0 + i*16 + 8*g + r;
                float val = acc[i][j][r] + bv;
                if constexpr (MODE == 0) {
                    Cf[(size_t)row * N + col] = val;
                } else if constexpr (MODE == 1) {
                    Ch[(size_t)row * N + col] = (_Float16)fmaxf(val, 0.0f);
                } else if constexpr (MODE == 2) {
                    const int s  = row >> 2;            // token row = s*B + b, B=4
                    const int hf = col >> 1;
                    const float ang = (float)s *
                        __expf((float)(2*hf) * (-9.2103403719761836f / 512.0f));
                    val += (col & 1) ? __cosf(ang) : __sinf(ang);
                    Cf[(size_t)row * N + col] = val;
                    Ch[(size_t)row * N + col] = (_Float16)val;
                } else { // MODE == 3 : QKV scatter
                    const int s   = row >> 2;
                    const int bb  = row & 3;
                    const int sec = col >> 9;           // 0=Q 1=K 2=V
                    const int cd  = col & 511;
                    const int h   = cd >> 6;
                    const int d   = cd & 63;
                    const size_t qk = (((size_t)(bb*H_ + h)) * S_ + s) * HD_ + d;
                    if (sec == 0)      Qh[qk] = (_Float16)(val * 0.125f); // 1/sqrt(64)
                    else if (sec == 1) Kh[qk] = (_Float16)val;
                    else Vt[(((size_t)(bb*H_ + h)) * HD_ + d) * S_ + s] = (_Float16)val;
                }
            }
        }
}

// ---------------------------------------------------------------------------
// Flash attention: grid (S/128, B*H), 8 waves/block, 16 queries per wave.
// Q[b,h,s,d] f16 (pre-scaled), K[b,h,s,d] f16, V^T[b,h,d,s] f16 -> Oh (S,B,D) f16.
// Online softmax over 64-key tiles. Softmax denominator computed with the
// matrix engine: wmma(P, ones) accumulates exact row sums into a C-layout
// fragment with the same (lane, r) -> row mapping as O. Only the row-max uses
// cross-lane shuffles. V fragments are loaded right after the score WMMAs and
// *pinned* there with a sched_barrier so their latency hides under the softmax
// VALU stretch; next K/V tiles are prefetched (global_prefetch_b8).
// ---------------------------------------------------------------------------
__global__ void __launch_bounds__(256)
attn_kernel(const _Float16* __restrict__ Qh, const _Float16* __restrict__ Kh,
            const _Float16* __restrict__ Vt, _Float16* __restrict__ Oh)
{
    __shared__ __attribute__((aligned(64))) _Float16 pbuf[8 * 16 * 72];

    const int lane = threadIdx.x & 31;
    const int wave = threadIdx.x >> 5;
    const int ll   = lane & 15;
    const int g    = lane >> 4;
    const int bh   = blockIdx.y;
    const int q0   = blockIdx.x * 128 + wave * 16;

    const _Float16* Qb = Qh + (size_t)bh * S_ * HD_;
    const _Float16* Kb = Kh + (size_t)bh * S_ * HD_;
    const _Float16* Vb = Vt + (size_t)bh * HD_ * S_;
    _Float16* pl = pbuf + wave * 16 * 72;

    // Resident Q A-fragments: 16 queries x 64 head-dims (two K=32 steps).
    FragH qa[2];
    #pragma unroll
    for (int kk = 0; kk < 2; kk++) {
        const _Float16* qp = Qb + (size_t)(q0 + ll) * HD_ + kk*32 + g*16;
        qa[kk].h[0] = *(const v8h*)qp;
        qa[kk].h[1] = *(const v8h*)(qp + 8);
    }

    // all-ones B fragment for row-sum WMMA
    v16h ones;
    #pragma unroll
    for (int e = 0; e < 16; e++) ones[e] = (_Float16)1.0f;

    v8f o[4];
    #pragma unroll
    for (int j = 0; j < 4; j++)
        #pragma unroll
        for (int e = 0; e < 8; e++) o[j][e] = 0.0f;
    v8f lacc;
    #pragma unroll
    for (int e = 0; e < 8; e++) lacc[e] = 0.0f;
    float mrun[8];
    #pragma unroll
    for (int r = 0; r < 8; r++) mrun[r] = -1e30f;

    for (int kb = 0; kb < S_; kb += 64) {
        // ---- batched K-fragment loads for this tile ----
        v16h kf[2][4];
        #pragma unroll
        for (int kk = 0; kk < 2; kk++)
            #pragma unroll
            for (int j = 0; j < 4; j++)
                kf[kk][j] = *(const v16h*)(Kb + (size_t)(kb + j*16 + ll) * HD_
                                              + kk*32 + g*16);

        // scores tile 16(q) x 64(k)
        v8f sf[4];
        #pragma unroll
        for (int j = 0; j < 4; j++) {
            v8f s;
            #pragma unroll
            for (int e = 0; e < 8; e++) s[e] = 0.0f;
            #pragma unroll
            for (int kk = 0; kk < 2; kk++)
                s = wmma_f16(qa[kk].v, kf[kk][j], s);
            sf[j] = s;
        }

        // ---- issue V-fragment loads now; latency hides under softmax ----
        v16h vf[2][4];
        #pragma unroll
        for (int kk = 0; kk < 2; kk++)
            #pragma unroll
            for (int j = 0; j < 4; j++)
                vf[kk][j] = *(const v16h*)(Vb + (size_t)(j*16 + ll) * S_
                                              + kb + kk*32 + g*16);

        // ---- prefetch next tile (global_prefetch_b8) ----
        if (kb + 64 < S_) {
            const int kbn = kb + 64;
            __builtin_prefetch(Kb + (size_t)(kbn + lane) * HD_, 0, 3);
            __builtin_prefetch(Kb + (size_t)(kbn + 32 + lane) * HD_, 0, 3);
            __builtin_prefetch(Vb + (size_t)lane * S_ + kbn, 0, 3);
            __builtin_prefetch(Vb + (size_t)(lane + 32) * S_ + kbn, 0, 3);
        }
        sched_fence();   // pin V loads + prefetches above the softmax VALU work

        // ---- online softmax: row m = 8*g + r across 16 lanes of a half ----
        float scl[8];
        #pragma unroll
        for (int r = 0; r < 8; r++) {
            float mx = fmaxf(fmaxf(sf[0][r], sf[1][r]),
                             fmaxf(sf[2][r], sf[3][r]));
            #pragma unroll
            for (int d = 1; d < 16; d <<= 1)
                mx = fmaxf(mx, __shfl_xor(mx, d, 32));
            const float mnew = fmaxf(mrun[r], mx);
            scl[r] = __expf(mrun[r] - mnew);
            mrun[r] = mnew;
            #pragma unroll
            for (int j = 0; j < 4; j++)
                sf[j][r] = __expf(sf[j][r] - mnew);
        }
        #pragma unroll
        for (int j = 0; j < 4; j++)
            #pragma unroll
            for (int r = 0; r < 8; r++) o[j][r] *= scl[r];
        #pragma unroll
        for (int r = 0; r < 8; r++) lacc[r] *= scl[r];

        // ---- C-layout P -> LDS (f16), read back as A-fragments ----
        // Same-wave DS ops are in-order: no barrier needed.
        #pragma unroll
        for (int j = 0; j < 4; j++)
            #pragma unroll
            for (int r = 0; r < 8; r++)
                pl[(8*g + r) * 72 + j*16 + ll] = (_Float16)sf[j][r];

        #pragma unroll
        for (int kk = 0; kk < 2; kk++) {
            FragH pa;
            const _Float16* pp = pl + ll * 72 + kk*32 + g*16;
            pa.h[0] = *(const v8h*)pp;
            pa.h[1] = *(const v8h*)(pp + 8);
            #pragma unroll
            for (int j = 0; j < 4; j++)
                o[j] = wmma_f16(pa.v, vf[kk][j], o[j]);
            // row-sum of P via matrix engine: denominator accumulator
            lacc = wmma_f16(pa.v, ones, lacc);
        }
    }

    // normalize + store to (S,B,D) f16
    const int b = bh >> 3, h = bh & 7;
    #pragma unroll
    for (int r = 0; r < 8; r++) {
        const float inv = 1.0f / lacc[r];
        const int s = q0 + 8*g + r;
        #pragma unroll
        for (int j = 0; j < 4; j++)
            Oh[((size_t)(s*B_ + b)) * D_ + h*HD_ + j*16 + ll] =
                (_Float16)(o[j][r] * inv);
    }
}

// ---------------------------------------------------------------------------
// LayerNorm(x + delta): one wave per 512-wide row. Writes f32 + f16.
// Safe to alias xo == x (each thread reads its elements before writing them).
// ---------------------------------------------------------------------------
__global__ void __launch_bounds__(256)
ln_kernel(const float* __restrict__ x, const float* __restrict__ dl,
          const float* __restrict__ gg, const float* __restrict__ bb,
          float* __restrict__ xo, _Float16* __restrict__ ho)
{
    const int lane = threadIdx.x & 31;
    const int wave = threadIdx.x >> 5;
    const int row  = blockIdx.x * 8 + wave;
    const float* xr = x  + (size_t)row * D_;
    const float* dr = dl + (size_t)row * D_;

    float v[16], s = 0.0f, sq = 0.0f;
    #pragma unroll
    for (int i = 0; i < 16; i++) {
        const float t = xr[i*32 + lane] + dr[i*32 + lane];
        v[i] = t; s += t; sq += t * t;
    }
    #pragma unroll
    for (int d = 1; d < 32; d <<= 1) {
        s  += __shfl_xor(s,  d, 32);
        sq += __shfl_xor(sq, d, 32);
    }
    const float mean = s  * (1.0f / 512.0f);
    const float var  = sq * (1.0f / 512.0f) - mean * mean;
    const float rstd = rsqrtf(var + 1e-5f);
    #pragma unroll
    for (int i = 0; i < 16; i++) {
        const int c = i*32 + lane;
        const float y = (v[i] - mean) * rstd * gg[c] + bb[c];
        xo[(size_t)row * D_ + c] = y;
        ho[(size_t)row * D_ + c] = (_Float16)y;
    }
}

// ---------------------------------------------------------------------------
__global__ void f2h_kernel(const float* __restrict__ src,
                           _Float16* __restrict__ dst, int n)
{
    const int i = blockIdx.x * blockDim.x + threadIdx.x;
    if (i < n) dst[i] = (_Float16)src[i];
}

// ---------------------------------------------------------------------------
extern "C" void kernel_launch(void* const* d_in, const int* in_sizes, int n_in,
                              void* d_out, int out_size, void* d_ws, size_t ws_size,
                              hipStream_t stream)
{
    const float* src   = (const float*)d_in[0];
    const float* w_in  = (const float*)d_in[1];
    const float* b_in  = (const float*)d_in[2];
    const float* in_w  = (const float*)d_in[3];
    const float* in_b  = (const float*)d_in[4];
    const float* out_w = (const float*)d_in[5];
    const float* out_b = (const float*)d_in[6];
    const float* w1    = (const float*)d_in[7];
    const float* b1    = (const float*)d_in[8];
    const float* w2    = (const float*)d_in[9];
    const float* b2    = (const float*)d_in[10];
    const float* ln1g  = (const float*)d_in[11];
    const float* ln1b  = (const float*)d_in[12];
    const float* ln2g  = (const float*)d_in[13];
    const float* ln2b  = (const float*)d_in[14];
    const float* w_out = (const float*)d_in[15];
    const float* b_out = (const float*)d_in[16];

    char* bump = (char*)d_ws;
    auto alloc = [&](size_t bytes) -> char* {
        char* r = bump;
        bump += (bytes + 255) & ~(size_t)255;
        return r;
    };

    _Float16* srch  = (_Float16*)alloc((size_t)MTOK * IND_ * 2);
    _Float16* winh  = (_Float16*)alloc((size_t)D_ * IND_ * 2);
    _Float16* inwh  = (_Float16*)alloc((size_t)L_ * 3*D_ * D_ * 2);
    _Float16* outwh = (_Float16*)alloc((size_t)L_ * D_ * D_ * 2);
    _Float16* w1h   = (_Float16*)alloc((size_t)L_ * FF_ * D_ * 2);
    _Float16* w2h   = (_Float16*)alloc((size_t)L_ * D_ * FF_ * 2);
    _Float16* wouth = (_Float16*)alloc((size_t)OUTD_ * D_ * 2);
    float*    x     = (float*)   alloc((size_t)MTOK * D_ * 4);
    _Float16* xh    = (_Float16*)alloc((size_t)MTOK * D_ * 2);
    float*    delta = (float*)   alloc((size_t)MTOK * D_ * 4);
    _Float16* Qp    = (_Float16*)alloc((size_t)B_ * H_ * S_ * HD_ * 2);
    _Float16* Kp    = (_Float16*)alloc((size_t)B_ * H_ * S_ * HD_ * 2);
    _Float16* Vp    = (_Float16*)alloc((size_t)B_ * H_ * HD_ * S_ * 2);
    _Float16* Oh    = (_Float16*)alloc((size_t)MTOK * D_ * 2);
    _Float16* hh    = (_Float16*)alloc((size_t)MTOK * FF_ * 2);

    auto cvt = [&](const float* s, _Float16* d, int n) {
        f2h_kernel<<<(n + 255) / 256, 256, 0, stream>>>(s, d, n);
    };
    cvt(src,   srch,  MTOK * IND_);
    cvt(w_in,  winh,  D_ * IND_);
    cvt(in_w,  inwh,  L_ * 3*D_ * D_);
    cvt(out_w, outwh, L_ * D_ * D_);
    cvt(w1,    w1h,   L_ * FF_ * D_);
    cvt(w2,    w2h,   L_ * D_ * FF_);
    cvt(w_out, wouth, OUTD_ * D_);

    const dim3 blk(256);
    // embed: x = src*w_in^T + b_in + posenc  (M=8192, N=512, K=64)
    gemm_kernel<2><<<dim3(MTOK/256, D_/64), blk, 0, stream>>>(
        srch, winh, b_in, x, xh, nullptr, nullptr, nullptr, D_, IND_);

    for (int l = 0; l < L_; l++) {
        // QKV: N=1536, K=512 -> scatter to Q/K/V^T
        gemm_kernel<3><<<dim3(MTOK/256, (3*D_)/64), blk, 0, stream>>>(
            xh, inwh + (size_t)l * 3*D_ * D_, in_b + (size_t)l * 3*D_,
            nullptr, nullptr, Qp, Kp, Vp, 3*D_, D_);

        // flash attention -> Oh (S,B,D) f16
        attn_kernel<<<dim3(S_/128, B_*H_), blk, 0, stream>>>(Qp, Kp, Vp, Oh);

        // out-proj: delta = Oh*out_w^T + out_b  (N=512, K=512)
        gemm_kernel<0><<<dim3(MTOK/256, D_/64), blk, 0, stream>>>(
            Oh, outwh + (size_t)l * D_ * D_, out_b + (size_t)l * D_,
            delta, nullptr, nullptr, nullptr, nullptr, D_, D_);

        // x = LN1(x + delta)
        ln_kernel<<<MTOK/8, blk, 0, stream>>>(
            x, delta, ln1g + (size_t)l * D_, ln1b + (size_t)l * D_, x, xh);

        // FF1: hh = relu(x*w1^T + b1) f16  (N=2048, K=512)
        gemm_kernel<1><<<dim3(MTOK/256, FF_/64), blk, 0, stream>>>(
            xh, w1h + (size_t)l * FF_ * D_, b1 + (size_t)l * FF_,
            nullptr, hh, nullptr, nullptr, nullptr, FF_, D_);

        // FF2: delta = hh*w2^T + b2  (N=512, K=2048)
        gemm_kernel<0><<<dim3(MTOK/256, D_/64), blk, 0, stream>>>(
            hh, w2h + (size_t)l * D_ * FF_, b2 + (size_t)l * D_,
            delta, nullptr, nullptr, nullptr, nullptr, D_, FF_);

        // x = LN2(x + delta)
        ln_kernel<<<MTOK/8, blk, 0, stream>>>(
            x, delta, ln2g + (size_t)l * D_, ln2b + (size_t)l * D_, x, xh);
    }

    // out = x*w_out^T + b_out  (N=64, K=512) -> f32 d_out
    gemm_kernel<0><<<dim3(MTOK/256, OUTD_/64), blk, 0, stream>>>(
        xh, wouth, b_out, (float*)d_out, nullptr, nullptr, nullptr, nullptr,
        OUTD_, D_);

    (void)in_sizes; (void)n_in; (void)out_size; (void)ws_size;
}
